// BoundaryLoss_35562329211067
// MI455X (gfx1250) — compile-verified, matrix operand order checked by reading, not run on hardware
//
#include <hip/hip_runtime.h>
#include <hip/hip_bf16.h>

typedef __attribute__((ext_vector_type(16))) _Float16 v16h;
typedef __attribute__((ext_vector_type(8)))  float    v8f;

#define HDIM 256
#define WDIM 256
#define EDT_LARGE 1e8f

// ---------------------------------------------------------------------------
// Kernel 1: argmax over C (first-max, matches jnp.argmax tie rule), copy both
// masks to uint8, and record per-(src,b,class) presence flags.
// ---------------------------------------------------------------------------
__global__ void prep_kernel(const float* __restrict__ pred,
                            const int* __restrict__ target,
                            unsigned char* __restrict__ masks, // [2][N]
                            int* __restrict__ flags,           // [2][B][C]
                            int B, int C, int N) {
  int i = blockIdx.x * blockDim.x + threadIdx.x;
  if (i >= N) return;
  const int HW = HDIM * WDIM;
  int b  = i / HW;
  int hw = i - b * HW;
  const float* p = pred + (size_t)b * C * HW + hw;
  float best = p[0];
  int bc = 0;
  for (int c = 1; c < C; ++c) {
    float v = p[(size_t)c * HW];
    if (v > best) { best = v; bc = c; }
  }
  int tc = target[i];
  masks[i]     = (unsigned char)bc;
  masks[N + i] = (unsigned char)tc;
  atomicOr(&flags[(0 * B + b) * C + bc], 1);
  atomicOr(&flags[(1 * B + b) * C + tc], 1);
}

// ---------------------------------------------------------------------------
// Kernel 2: 1D EDT along W. f in {0, LARGE} -> g(x) = min over seeds of
// (x-y)^2, or LARGE if the row has no seed (identical to the reference's
// min-plus: a seed candidate (<=255^2) always beats 1e8, and an empty row's
// reference min is 1e8 + 0 at y==x). One block per row; the row's class ids
// live in LDS, each lane scans 256 candidates with fmin.
// ---------------------------------------------------------------------------
__global__ void row_edt_kernel(const unsigned char* __restrict__ masks,
                               float* __restrict__ g, int cls) {
  __shared__ int sm[WDIM];
  const int x = threadIdx.x;
  const size_t rowOff = (size_t)blockIdx.x * WDIM; // blocks = 2*B*H rows
  sm[x] = masks[rowOff + x];
  __syncthreads();
  float best = EDT_LARGE;
#pragma unroll 8
  for (int y = 0; y < WDIM; ++y) {
    int dx = x - y;
    float cand = (sm[y] == cls) ? (float)(dx * dx) : EDT_LARGE;
    best = fminf(best, cand);
  }
  g[rowOff + x] = best;
}

// ---------------------------------------------------------------------------
// Kernel 3: exact min-plus along H on general f32 values (up to 1e8), then
// sqrt * presence, accumulate into dt. One block = 16 columns x 256 rows of
// one (src,b) image. Tile in LDS (256x16 f32 = 16KB of the 320KB WGP LDS);
// each lane register-blocks 16 outputs so the column value gs[y][col] is
// read from LDS once per y (16 banks x 2-lane broadcast on 64 banks ->
// conflict-free).
// ---------------------------------------------------------------------------
__global__ void col_edt_kernel(const float* __restrict__ g,
                               const int* __restrict__ flags,
                               float* __restrict__ dt,
                               int B, int C, int cls) {
  __shared__ float gs[HDIM * 16];
  const int t = threadIdx.x;
  const int tilesPerImg = WDIM / 16;
  int tile = blockIdx.x % tilesPerImg;
  int img  = blockIdx.x / tilesPerImg;  // src*B + b
  int src  = img / B;
  int b    = img % B;
  int x0   = tile * 16;
  size_t base = (size_t)img * HDIM * WDIM;

#pragma unroll
  for (int i = 0; i < 16; ++i) {
    int idx = i * 256 + t;
    int row = idx >> 4;
    int col = idx & 15;
    gs[idx] = g[base + (size_t)row * WDIM + x0 + col];
  }
  __syncthreads();

  const int col   = t & 15;
  const int rbase = t >> 4;
  float best[16];
#pragma unroll
  for (int k = 0; k < 16; ++k) best[k] = EDT_LARGE;

  for (int y = 0; y < HDIM; ++y) {
    float v = gs[y * 16 + col];
    int dy = rbase - y;
#pragma unroll
    for (int k = 0; k < 16; ++k) {
      int d = dy + 16 * k;
      best[k] = fminf(best[k], v + (float)(d * d));
    }
  }

  float flagf = flags[(src * B + b) * C + cls] ? 1.0f : 0.0f;
#pragma unroll
  for (int k = 0; k < 16; ++k) {
    int y = rbase + 16 * k;
    dt[base + (size_t)y * WDIM + x0 + col] += sqrtf(best[k]) * flagf;
  }
}

// ---------------------------------------------------------------------------
// Kernel 4: sum |dt_pred - dt_target| with the matrix pipe. Each wave packs
// 512 absdiffs per iteration into a 16x32 f16 A-matrix; B = all-ones, so
// v_wmma_f32_16x16x32_f16 accumulates the sum into an f32 C (each value is
// counted 16x across D -> scale by 1/16). The f16 quantization error is
// carried exactly as an f32 residual on the VALU, so the total matches a
// pure-f32 reduction. Loop control goes through readfirstlane so the trip
// count lives in SGPRs: uniform s_cbranch flow, EXEC stays all-1s at every
// WMMA issue (hard ISA requirement), and no per-lane exec-mask churn.
// Per-wave partials -> workspace (no float atomics => deterministic).
// ---------------------------------------------------------------------------
__global__ void reduce_wmma_kernel(const float* __restrict__ dtp,
                                   const float* __restrict__ dtt,
                                   float* __restrict__ partials,
                                   int N, int numWaves) {
  int gtid = blockIdx.x * blockDim.x + threadIdx.x;
  int wave = __builtin_amdgcn_readfirstlane(gtid >> 5);  // scalarize: uniform
  int lane = gtid & 31;
  const int stride = numWaves * 512;

  v16h ones;
#pragma unroll
  for (int j = 0; j < 16; ++j) ones[j] = (_Float16)1.0f;

  v8f acc = {};
  float resid = 0.0f;

  for (int base = wave * 512; base < N; base += stride) {
    int e0 = base + lane * 16;
    v16h a;
    if (base + 512 <= N) {  // scalar (SGPR) condition -> uniform branch
      const float4* p4 = reinterpret_cast<const float4*>(dtp + e0);
      const float4* t4 = reinterpret_cast<const float4*>(dtt + e0);
#pragma unroll
      for (int q = 0; q < 4; ++q) {
        float4 pv = p4[q];
        float4 tv = t4[q];
        float v0 = fabsf(pv.x - tv.x);
        float v1 = fabsf(pv.y - tv.y);
        float v2 = fabsf(pv.z - tv.z);
        float v3 = fabsf(pv.w - tv.w);
        _Float16 h0 = (_Float16)v0, h1 = (_Float16)v1;
        _Float16 h2 = (_Float16)v2, h3 = (_Float16)v3;
        a[4 * q + 0] = h0; a[4 * q + 1] = h1;
        a[4 * q + 2] = h2; a[4 * q + 3] = h3;
        resid += (v0 - (float)h0) + (v1 - (float)h1) +
                 (v2 - (float)h2) + (v3 - (float)h3);
      }
    } else {
#pragma unroll
      for (int j = 0; j < 16; ++j) {
        int e = e0 + j;
        float v = (e < N) ? fabsf(dtp[e] - dtt[e]) : 0.0f;  // select, no branch
        _Float16 h = (_Float16)v;
        a[j] = h;
        resid += v - (float)h;
      }
    }
    int nb = base + stride;
    if (nb < N) {  // uniform; gfx1250 speculative prefetch of next chunk
      __builtin_prefetch(dtp + nb + lane * 16, 0, 0);
      __builtin_prefetch(dtt + nb + lane * 16, 0, 0);
    }
    acc = __builtin_amdgcn_wmma_f32_16x16x32_f16(
        /*neg_a=*/false, a, /*neg_b=*/false, ones,
        /*c_mod=*/(short)0, acc, /*reuse_a=*/false, /*reuse_b=*/false);
  }

  float s = acc[0] + acc[1] + acc[2] + acc[3] +
            acc[4] + acc[5] + acc[6] + acc[7];
  float tsum = s * (1.0f / 16.0f) + resid;
#pragma unroll
  for (int off = 16; off > 0; off >>= 1)
    tsum += __shfl_down(tsum, off, 32);
  if (lane == 0) partials[wave] = tsum;
}

// Fixed-order final sum on a single wave -> deterministic output.
__global__ void final_kernel(const float* __restrict__ partials,
                             float* __restrict__ out, int numWaves, float invN) {
  int lane = threadIdx.x;  // 32 threads
  float s = 0.0f;
  for (int i = lane; i < numWaves; i += 32) s += partials[i];
#pragma unroll
  for (int off = 16; off > 0; off >>= 1)
    s += __shfl_down(s, off, 32);
  if (lane == 0) out[0] = s * invN;
}

// ---------------------------------------------------------------------------
static inline size_t align256(size_t x) { return (x + 255) & ~(size_t)255; }

extern "C" void kernel_launch(void* const* d_in, const int* in_sizes, int n_in,
                              void* d_out, int out_size, void* d_ws, size_t ws_size,
                              hipStream_t stream) {
  const float* pred  = (const float*)d_in[0];
  const int* target  = (const int*)d_in[1];

  const int N = in_sizes[1];                 // B*H*W = 262144
  const int C = in_sizes[0] / N;             // 4
  const int B = N / (HDIM * WDIM);           // 4

  const int reduceBlocks = 32;
  const int numWaves = reduceBlocks * 256 / 32;  // 256 waves

  // Workspace layout
  char* ws = (char*)d_ws;
  size_t off = 0;
  int*           flags    = (int*)(ws + off);           off += align256((size_t)2 * B * C * sizeof(int));
  float*         partials = (float*)(ws + off);         off += align256((size_t)numWaves * sizeof(float));
  unsigned char* masks    = (unsigned char*)(ws + off); off += align256((size_t)2 * N);
  float*         g        = (float*)(ws + off);         off += align256((size_t)2 * N * sizeof(float));
  float*         dt       = (float*)(ws + off);         off += align256((size_t)2 * N * sizeof(float));
  (void)ws_size; (void)n_in; (void)out_size;

  hipMemsetAsync(flags, 0, (size_t)2 * B * C * sizeof(int), stream);
  hipMemsetAsync(dt,    0, (size_t)2 * N * sizeof(float),   stream);

  prep_kernel<<<(N + 255) / 256, 256, 0, stream>>>(pred, target, masks, flags, B, C, N);

  for (int c = 1; c < C; ++c) {
    row_edt_kernel<<<2 * B * HDIM, 256, 0, stream>>>(masks, g, c);
    col_edt_kernel<<<2 * B * (WDIM / 16), 256, 0, stream>>>(g, flags, dt, B, C, c);
  }

  reduce_wmma_kernel<<<reduceBlocks, 256, 0, stream>>>(dt, dt + N, partials, N, numWaves);
  final_kernel<<<1, 32, 0, stream>>>(partials, (float*)d_out, numWaves, 1.0f / (float)N);
}
